// UpBlock_14070312862111
// MI455X (gfx1250) — compile-verified, matrix operand order checked by reading, not run on hardware
//
#include <hip/hip_runtime.h>

// ---------------------------------------------------------------------------
// CDNA5 / gfx1250 sparse-conv UpBlock.
// bf16 gather-GEMM via v_wmma_f32_16x16x32_bf16 (f32 accumulate), fp32 BN.
// All feature tensors fit in the 192MB L2 -> WMMA / L2-gather bound; bf16
// halves gather bytes and uses the 16x16x32 matrix pipe (vs 16x16x4 f32).
// Templated CIN/KOFF/MODE (fully unrolled K loop), 32-row M-tiles (B-fragment
// reuse x2, half the barriers), wave-ballot row-validity, and ASYNC
// global->LDS gather (ASYNCcnt + s_wait_asynccnt) on the gfx1250 path.
// ---------------------------------------------------------------------------

typedef __attribute__((ext_vector_type(16))) __bf16 v16bf;
typedef __attribute__((ext_vector_type(8)))  float  v8f;
typedef __attribute__((ext_vector_type(4)))  unsigned int uint4v;
typedef __attribute__((__vector_size__(4 * sizeof(int)))) int vsi4;

#define AS1 __attribute__((address_space(1)))
#define AS3 __attribute__((address_space(3)))

#if __has_builtin(__builtin_amdgcn_global_load_async_to_lds_b128) && \
    __has_builtin(__builtin_amdgcn_s_wait_asynccnt)
#define USE_ASYNC_GATHER 1
#else
#define USE_ASYNC_GATHER 0
#endif

struct Q2 { uint4v a, b; };   // 32 bytes == v16bf

__device__ __forceinline__ unsigned short f2bf(float f) {
    unsigned int u = __builtin_bit_cast(unsigned int, f);
    unsigned int r = (u + 0x7FFFu + ((u >> 16) & 1u)) >> 16;  // RNE
    return (unsigned short)r;
}

// ---------------------------------------------------------------------------
// elementwise f32 -> bf16 cast
// ---------------------------------------------------------------------------
__global__ void cast_bf16_kernel(const float* __restrict__ in,
                                 unsigned short* __restrict__ out, long n) {
    long i = (long)blockIdx.x * 256 + threadIdx.x;
    if (i < n) out[i] = f2bf(in[i]);
}

// ---------------------------------------------------------------------------
// weight cast + transpose: W[k][c][n] f32  ->  WT[k][n][c] bf16
// ---------------------------------------------------------------------------
__global__ void wcast_kernel(const float* __restrict__ w,
                             unsigned short* __restrict__ out,
                             int K, int CIN_) {
    long i = (long)blockIdx.x * 256 + threadIdx.x;
    long total = (long)K * CIN_ * 128;
    if (i >= total) return;
    int c = (int)(i % CIN_);
    long t = i / CIN_;
    int n = (int)(t % 128);
    int k = (int)(t / 128);
    out[i] = f2bf(w[((long)k * CIN_ + c) * 128 + n]);
}

// ---------------------------------------------------------------------------
// Gather-GEMM sparse conv.  COUT fixed at 128.
//  block = 256 threads = 8 wave32; block computes 32 output rows x 128 cols.
//  wave w owns N-tile [16w,16w+16) for both 16-row M-tiles (2 accumulators,
//  B fragment reused by both WMMAs).
//  MODE 0: out = leaky_relu(conv)  -> out_f32
//  MODE 1: out = conv + skip       -> out_bf16
// ---------------------------------------------------------------------------
template <int CIN_, int KOFF, int MODE>
__global__ void __launch_bounds__(256)
sconv_wmma_kernel(const unsigned short* __restrict__ feats, // [Nin, CIN_] bf16
                  const unsigned short* __restrict__ wt,    // [K][128][CIN_] bf16
                  const int*            __restrict__ idx,   // [K, Nout]
                  const float*          __restrict__ skip,  // MODE 1 only
                  float*                __restrict__ out_f32,
                  unsigned short*       __restrict__ out_bf16,
                  int Nout) {
    constexpr int ROWS = 32;
    constexpr int CHUNKS_PER_ROW = CIN_ >> 3;              // 16B chunks per row
    constexpr int TOTAL_CHUNKS   = ROWS * CHUNKS_PER_ROW;  // 512 or 1024

    __shared__ __align__(16) unsigned short sA[ROWS * CIN_];
    __shared__ int sIdx[ROWS];
    __shared__ int sAny;

    const int t    = threadIdx.x;
    const int lane = t & 31;
    const int wave = t >> 5;
    const int row0 = blockIdx.x * ROWS;
    const int n0   = wave * 16;
    const int col  = lane & 15;
    const int ko   = (lane & 16) ? 8 : 0;  // K sub-offset per ISA A/B lane layout

    v8f acc0 = {0.f, 0.f, 0.f, 0.f, 0.f, 0.f, 0.f, 0.f};
    v8f acc1 = {0.f, 0.f, 0.f, 0.f, 0.f, 0.f, 0.f, 0.f};

    for (int k = 0; k < KOFF; ++k) {
        __syncthreads();                 // previous MMA finished reading sA
        bool ok = false;
        if (t < ROWS) {                  // loader lanes == wave 0
            int r  = row0 + t;
            int my = (r < Nout) ? idx[(long)k * Nout + r] : -1;
            sIdx[t] = my;
            ok = (my >= 0);
        }
        unsigned long long m = __ballot(ok);
        if (t == 0) sAny = (m != 0ull) ? 1 : 0;
        __syncthreads();

        const bool any = (sAny != 0);    // uniform across block
        if (any) {
            // cooperative gather of 32 rows into LDS (zeros for missing rows)
            #pragma unroll
            for (int c = 0; c < TOTAL_CHUNKS / 256; ++c) {
                int ch = t + c * 256;
                int r  = ch / CHUNKS_PER_ROW;          // constant divisor
                int co = ch & (CHUNKS_PER_ROW - 1);
                int src = sIdx[r];
                unsigned short* ldst = sA + r * CIN_ + co * 8;
                if (src >= 0) {
#if USE_ASYNC_GATHER
                    __builtin_amdgcn_global_load_async_to_lds_b128(
                        (AS1 vsi4*)(feats + (long)src * CIN_ + co * 8),
                        (AS3 vsi4*)ldst, 0, 0);
#else
                    *(uint4v*)ldst =
                        *(const uint4v*)(feats + (long)src * CIN_ + co * 8);
#endif
                } else {
                    uint4v z = {0u, 0u, 0u, 0u};
                    *(uint4v*)ldst = z;
                }
            }
        }
#if USE_ASYNC_GATHER
        __builtin_amdgcn_s_wait_asynccnt(0);
#endif
        __syncthreads();

        if (any) {   // uniform -> EXEC all-ones for WMMA
            const unsigned short* wb = wt + ((long)k * 128 + n0 + col) * CIN_;
            const unsigned short* a0 = sA + (lane & 15) * CIN_ + ko;
            const unsigned short* a1 = a0 + 16 * CIN_;
            #pragma unroll
            for (int kc = 0; kc < CIN_; kc += 32) {
                Q2 qb, qa0, qa1;
                qb.a  = *(const uint4v*)(wb + kc + ko);
                qb.b  = *(const uint4v*)(wb + kc + ko + 16);
                qa0.a = *(const uint4v*)(a0 + kc);
                qa0.b = *(const uint4v*)(a0 + kc + 16);
                qa1.a = *(const uint4v*)(a1 + kc);
                qa1.b = *(const uint4v*)(a1 + kc + 16);
                v16bf bv  = __builtin_bit_cast(v16bf, qb);
                v16bf av0 = __builtin_bit_cast(v16bf, qa0);
                v16bf av1 = __builtin_bit_cast(v16bf, qa1);
                acc0 = __builtin_amdgcn_wmma_f32_16x16x32_bf16(
                           false, av0, false, bv, (short)0, acc0, false, false);
                acc1 = __builtin_amdgcn_wmma_f32_16x16x32_bf16(
                           false, av1, false, bv, (short)0, acc1, false, false);
            }
        }
    }

    // Epilogue.  D layout: lanes 0-15 -> N=lane, M=r (0..7);
    //            lanes 16-31 -> N=lane-16, M=8+r.
    const int mbase = (lane & 16) ? 8 : 0;
    #pragma unroll
    for (int r = 0; r < 8; ++r) {
        int rowA = row0 + mbase + r;
        int rowB = rowA + 16;
        if (rowA < Nout) {
            long o = (long)rowA * 128 + n0 + col;
            float v = acc0[r];
            if (MODE == 0) {
                v = (v >= 0.f) ? v : 0.01f * v;
                out_f32[o] = v;
            } else {
                v += skip[o];
                out_bf16[o] = f2bf(v);
            }
        }
        if (rowB < Nout) {
            long o = (long)rowB * 128 + n0 + col;
            float v = acc1[r];
            if (MODE == 0) {
                v = (v >= 0.f) ? v : 0.01f * v;
                out_f32[o] = v;
            } else {
                v += skip[o];
                out_bf16[o] = f2bf(v);
            }
        }
    }
}

// ---------------------------------------------------------------------------
// BatchNorm helpers (training-mode stats over all active voxels)
// ---------------------------------------------------------------------------
__global__ void zero_stats_kernel(float* p) { p[threadIdx.x] = 0.f; } // 256 thr

__global__ void __launch_bounds__(128)
bn_reduce_kernel(const float* __restrict__ x, int N, float* __restrict__ stats) {
    const int c = threadIdx.x;                 // channel 0..127
    const long r0 = (long)blockIdx.x * 256;
    float s = 0.f, s2 = 0.f;
    for (int i = 0; i < 256; ++i) {
        long r = r0 + i;
        if (r < N) {
            float v = x[r * 128 + c];
            s += v; s2 += v * v;
        }
    }
    atomicAdd(&stats[c], s);
    atomicAdd(&stats[128 + c], s2);
}

// MODE 0: f32 out   MODE 1: bf16 out
template <int MODE>
__global__ void bn_apply_kernel(const float* __restrict__ x,
                                const float* __restrict__ stats,
                                const float* __restrict__ gamma,
                                const float* __restrict__ beta,
                                long total, float inv_n,
                                float* __restrict__ out_f32,
                                unsigned short* __restrict__ out_bf16) {
    long i = (long)blockIdx.x * 256 + threadIdx.x;
    if (i >= total) return;
    int c = (int)(i & 127);
    float m  = stats[c] * inv_n;
    float vv = stats[128 + c] * inv_n - m * m;
    float sc = rsqrtf(vv + 1e-3f) * gamma[c];
    float y  = (x[i] - m) * sc + beta[c];
    if (MODE == 0) out_f32[i] = y;
    else           out_bf16[i] = f2bf(y);
}

// ---------------------------------------------------------------------------
// host-side orchestration
// ---------------------------------------------------------------------------
extern "C" void kernel_launch(void* const* d_in, const int* in_sizes, int n_in,
                              void* d_out, int out_size, void* d_ws, size_t ws_size,
                              hipStream_t stream) {
    (void)n_in; (void)out_size; (void)ws_size;

    const float* x_feats = (const float*)d_in[0];
    const float* skip    = (const float*)d_in[1];
    const float* W_trans = (const float*)d_in[2];
    const float* g_t     = (const float*)d_in[3];
    const float* b_t     = (const float*)d_in[4];
    const float* W_up    = (const float*)d_in[5];
    const float* W1      = (const float*)d_in[6];
    const float* g1      = (const float*)d_in[7];
    const float* b1      = (const float*)d_in[8];
    const float* W2      = (const float*)d_in[9];
    const float* g2      = (const float*)d_in[10];
    const float* b2      = (const float*)d_in[11];
    const float* W3      = (const float*)d_in[12];
    const float* g3      = (const float*)d_in[13];
    const float* b3      = (const float*)d_in[14];
    const int* idx_trans = (const int*)d_in[15];
    const int* idx_up    = (const int*)d_in[16];
    const int* idx_c1    = (const int*)d_in[17];
    const int* idx_c2    = (const int*)d_in[18];
    const int* idx_c3    = (const int*)d_in[19];

    const int CIN = 256, COUT = 128;
    const int NC = in_sizes[0] / CIN;   // 60000
    const int NF = in_sizes[1] / COUT;  // 200000

    char* ws = (char*)d_ws;
    size_t off = 0;
    auto carve = [&](size_t bytes) -> char* {
        char* p = ws + off;
        off = (off + bytes + 255) & ~(size_t)255;
        return p;
    };
    unsigned short* XB  = (unsigned short*)carve((size_t)NC * CIN * 2);
    unsigned short* WTt = (unsigned short*)carve((size_t)27 * COUT * CIN * 2);
    unsigned short* WTu = (unsigned short*)carve((size_t)27 * COUT * COUT * 2);
    unsigned short* WT1 = (unsigned short*)carve((size_t)9  * COUT * COUT * 2);
    unsigned short* WT2 = (unsigned short*)carve((size_t)9  * COUT * COUT * 2);
    unsigned short* WT3 = (unsigned short*)carve((size_t)27 * COUT * COUT * 2);
    float*          TCf = (float*)         carve((size_t)NC * COUT * 4);
    unsigned short* TCb = (unsigned short*)carve((size_t)NC * COUT * 2);
    unsigned short* FB  = (unsigned short*)carve((size_t)NF * COUT * 2);
    float*          F1f = (float*)         carve((size_t)NF * COUT * 4);
    float*          stats = (float*)       carve(256 * 4);
    float* OUT = (float*)d_out;

    auto gr1 = [](long n) { return (unsigned)((n + 255) / 256); };

    // 1. casts
    cast_bf16_kernel<<<gr1((long)NC * CIN), 256, 0, stream>>>(x_feats, XB, (long)NC * CIN);
    wcast_kernel<<<gr1((long)27 * CIN  * 128), 256, 0, stream>>>(W_trans, WTt, 27, CIN);
    wcast_kernel<<<gr1((long)27 * COUT * 128), 256, 0, stream>>>(W_up,    WTu, 27, COUT);
    wcast_kernel<<<gr1((long) 9 * COUT * 128), 256, 0, stream>>>(W1,      WT1,  9, COUT);
    wcast_kernel<<<gr1((long) 9 * COUT * 128), 256, 0, stream>>>(W2,      WT2,  9, COUT);
    wcast_kernel<<<gr1((long)27 * COUT * 128), 256, 0, stream>>>(W3,      WT3, 27, COUT);

    const unsigned tilesC = (unsigned)((NC + 31) / 32);
    const unsigned tilesF = (unsigned)((NF + 31) / 32);

    // 2. trans_dilao (27, 256->128) + leaky -> TCf ; BN -> TCb (bf16)
    sconv_wmma_kernel<256, 27, 0><<<tilesC, 256, 0, stream>>>(
        XB, WTt, idx_trans, nullptr, TCf, nullptr, NC);
    zero_stats_kernel<<<1, 256, 0, stream>>>(stats);
    bn_reduce_kernel<<<gr1((long)NC), 128, 0, stream>>>(TCf, NC, stats);
    bn_apply_kernel<1><<<gr1((long)NC * 128), 256, 0, stream>>>(
        TCf, stats, g_t, b_t, (long)NC * 128, 1.f / (float)NC, nullptr, TCb);

    // 3. up_subm (27, 128->128) + skip -> FB (bf16)
    sconv_wmma_kernel<128, 27, 1><<<tilesF, 256, 0, stream>>>(
        TCb, WTu, idx_up, skip, nullptr, FB, NF);

    // 4. conv1 (9) + leaky -> F1f ; BN -> FB
    sconv_wmma_kernel<128, 9, 0><<<tilesF, 256, 0, stream>>>(
        FB, WT1, idx_c1, nullptr, F1f, nullptr, NF);
    zero_stats_kernel<<<1, 256, 0, stream>>>(stats);
    bn_reduce_kernel<<<gr1((long)NF), 128, 0, stream>>>(F1f, NF, stats);
    bn_apply_kernel<1><<<gr1((long)NF * 128), 256, 0, stream>>>(
        F1f, stats, g1, b1, (long)NF * 128, 1.f / (float)NF, nullptr, FB);

    // 5. conv2 (9) + leaky -> F1f ; BN -> FB
    sconv_wmma_kernel<128, 9, 0><<<tilesF, 256, 0, stream>>>(
        FB, WT2, idx_c2, nullptr, F1f, nullptr, NF);
    zero_stats_kernel<<<1, 256, 0, stream>>>(stats);
    bn_reduce_kernel<<<gr1((long)NF), 128, 0, stream>>>(F1f, NF, stats);
    bn_apply_kernel<1><<<gr1((long)NF * 128), 256, 0, stream>>>(
        F1f, stats, g2, b2, (long)NF * 128, 1.f / (float)NF, nullptr, FB);

    // 6. conv3 (27) + leaky -> F1f ; BN -> d_out (f32)
    sconv_wmma_kernel<128, 27, 0><<<tilesF, 256, 0, stream>>>(
        FB, WT3, idx_c3, nullptr, F1f, nullptr, NF);
    zero_stats_kernel<<<1, 256, 0, stream>>>(stats);
    bn_reduce_kernel<<<gr1((long)NF), 128, 0, stream>>>(F1f, NF, stats);
    bn_apply_kernel<0><<<gr1((long)NF * 128), 256, 0, stream>>>(
        F1f, stats, g3, b3, (long)NF * 128, 1.f / (float)NF, OUT, nullptr);
}
